// Seq2SeqDialogModel_51333449122323
// MI455X (gfx1250) — matrix operand order repaired and checked by reading, not compiled, not run on hardware
//
#include <hip/hip_runtime.h>
#include <math.h>

typedef __bf16 bf16;
typedef __attribute__((ext_vector_type(16))) __bf16 v16bf;
typedef __attribute__((ext_vector_type(8)))  __bf16 v8bf;
typedef __attribute__((ext_vector_type(8)))  float   v8f;

#define DEVINL __device__ __forceinline__

// ---------------------------------------------------------------------------
// WMMA dual-N-tile inner loop: one A tile feeds two adjacent 16x16 output
// tiles (N-blocking x2) -> 6 b128 loads per 2 WMMAs instead of 8.
// A layout per ISA: lane<16 holds K {k..k+7, k+16..k+23}; lane>=16 the other 8s.
// B layout matches A: lane<16 holds K {k..k+7, k+16..k+23} of column n; lane>=16 the other 8s.
// ---------------------------------------------------------------------------
DEVINL void wmma_acc2(const bf16* __restrict__ Arow,
                      const bf16* __restrict__ Brow0,
                      const bf16* __restrict__ Brow1,
                      int K, int kh, v8f& acc0, v8f& acc1) {
  for (int k = 0; k < K; k += 32) {
    v16bf a, b0, b1;
    const bf16* ap = Arow + k + kh * 8;
    reinterpret_cast<v8bf*>(&a)[0] = *reinterpret_cast<const v8bf*>(ap);
    reinterpret_cast<v8bf*>(&a)[1] = *reinterpret_cast<const v8bf*>(ap + 16);
    const bf16* bp0 = Brow0 + k + kh * 8;
    reinterpret_cast<v8bf*>(&b0)[0] = *reinterpret_cast<const v8bf*>(bp0);
    reinterpret_cast<v8bf*>(&b0)[1] = *reinterpret_cast<const v8bf*>(bp0 + 16);
    const bf16* bp1 = Brow1 + k + kh * 8;
    reinterpret_cast<v8bf*>(&b1)[0] = *reinterpret_cast<const v8bf*>(bp1);
    reinterpret_cast<v8bf*>(&b1)[1] = *reinterpret_cast<const v8bf*>(bp1 + 16);
    acc0 = __builtin_amdgcn_wmma_f32_16x16x32_bf16(false, a, false, b0,
                                                   (short)0, acc0, false, false);
    acc1 = __builtin_amdgcn_wmma_f32_16x16x32_bf16(false, a, false, b1,
                                                   (short)0, acc1, false, false);
  }
}

// ---------------------------------------------------------------------------
// Generic GEMM: C[m,n] = act( sum_k A0[m,k]*B0[n,k] + sum_k A1[m,k]*B1[n,k]
//                             + bias[n] + addC[m,n] )
// One wave per 16x32 output block (two 16x16 WMMA tiles sharing A).
// M mult of 16; N mult of 32; K0,K1 mult of 32.
// ---------------------------------------------------------------------------
__global__ void gemm_bf16_kernel(const bf16* A0, int lda0, const bf16* B0, int ldb0, int K0,
                                 const bf16* A1, int lda1, const bf16* B1, int ldb1, int K1,
                                 const float* bias, const float* addC,
                                 float* C, bf16* Cbf, int M, int N, int act) {
  int lane = threadIdx.x & 31;
  int w = (blockIdx.x * blockDim.x + threadIdx.x) >> 5;
  int tNp = N >> 5, tM = M >> 4;
  if (w >= tM * tNp) return;
  int tm = w / tNp, tnp = w - tm * tNp;
  int kh = lane >> 4, rl = lane & 15;
  v8f acc0 = {}, acc1 = {};
  {
    const bf16* Ar = A0 + (size_t)(tm * 16 + rl) * lda0;
    const bf16* Br = B0 + (size_t)(tnp * 32 + rl) * ldb0;
    wmma_acc2(Ar, Br, Br + (size_t)16 * ldb0, K0, kh, acc0, acc1);
  }
  if (A1) {
    const bf16* Ar = A1 + (size_t)(tm * 16 + rl) * lda1;
    const bf16* Br = B1 + (size_t)(tnp * 32 + rl) * ldb1;
    wmma_acc2(Ar, Br, Br + (size_t)16 * ldb1, K1, kh, acc0, acc1);
  }
  int n0 = tnp * 32 + rl;
  int n1 = n0 + 16;
  int mb = tm * 16 + kh * 8;
  float bn0 = bias ? bias[n0] : 0.f;
  float bn1 = bias ? bias[n1] : 0.f;
#pragma unroll
  for (int r = 0; r < 8; ++r) {
    int m = mb + r;
    float v0 = acc0[r] + bn0;
    float v1 = acc1[r] + bn1;
    if (addC) { v0 += addC[(size_t)m * N + n0]; v1 += addC[(size_t)m * N + n1]; }
    if (act)  { v0 = tanhf(v0); v1 = tanhf(v1); }
    C[(size_t)m * N + n0] = v0;
    C[(size_t)m * N + n1] = v1;
    if (Cbf) { Cbf[(size_t)m * N + n0] = (bf16)v0; Cbf[(size_t)m * N + n1] = (bf16)v1; }
  }
}

// ---------------------------------------------------------------------------
// Encoder step: 4 fused GEMMs (gi_f, gh_f, gi_b, gh_b), each 64x1536, K=512.
// One wave per 16x32 output block -> 4 items * (4*48) = 768 waves.
// ---------------------------------------------------------------------------
struct Enc4 {
  const bf16* A[4]; int lda[4];
  const bf16* B[4];
  const float* bias[4];
  float* C[4];
};

__global__ void enc_step_gemm(Enc4 p) {
  int lane = threadIdx.x & 31;
  int w = (blockIdx.x * blockDim.x + threadIdx.x) >> 5;
  const int tNp = 48, per = 4 * tNp;   // tM=4, N=1536 -> 48 tile pairs
  int item = w / per;
  if (item >= 4) return;
  int lw = w - item * per;
  int tm = lw / tNp, tnp = lw - tm * tNp;
  int kh = lane >> 4, rl = lane & 15;
  v8f acc0 = {}, acc1 = {};
  {
    const bf16* Ar = p.A[item] + (size_t)(tm * 16 + rl) * p.lda[item];
    const bf16* Br = p.B[item] + (size_t)(tnp * 32 + rl) * 512;
    wmma_acc2(Ar, Br, Br + (size_t)16 * 512, 512, kh, acc0, acc1);
  }
  int n0 = tnp * 32 + rl;
  int n1 = n0 + 16;
  int mb = tm * 16 + kh * 8;
  float bn0 = p.bias[item][n0];
  float bn1 = p.bias[item][n1];
  float* C = p.C[item];
#pragma unroll
  for (int r = 0; r < 8; ++r) {
    C[(size_t)(mb + r) * 1536 + n0] = acc0[r] + bn0;
    C[(size_t)(mb + r) * 1536 + n1] = acc1[r] + bn1;
  }
}

// ---------------------------------------------------------------------------
// Elementwise helpers
// ---------------------------------------------------------------------------
DEVINL float sigm(float x) { return 1.f / (1.f + expf(-x)); }

__global__ void cvt_f32_bf16(const float* s, bf16* d, int n) {
  int i = blockIdx.x * blockDim.x + threadIdx.x;
  if (i < n) d[i] = (bf16)s[i];
}

// one block per token: gather embedding row to bf16 (+mask for encoder side)
__global__ void embed_gather(const int* ids, const float* emb, bf16* out,
                             float* mask, int E) {
  int tok = blockIdx.x;
  int idx = ids[tok];
  const float* src = emb + (size_t)idx * E;
  bf16* dst = out + (size_t)tok * E;
  for (int e = threadIdx.x; e < E; e += blockDim.x) dst[e] = (bf16)src[e];
  if (mask && threadIdx.x == 0) mask[tok] = (idx != 0) ? 1.f : 0.f;
}

// Encoder gate math for both directions; step in [0,63].
__global__ void enc_gate(const float* gi_f, const float* gh_f,
                         const float* gi_b, const float* gh_b,
                         float* h_f, float* h_b, bf16* utterin,
                         float* enc_out, const float* mask, int step) {
  int i = blockIdx.x * blockDim.x + threadIdx.x;   // 2*64*512
  if (i >= 2 * 64 * 512) return;
  int j = i & 511;
  int b = (i >> 9) & 63;
  int dir = i >> 15;
  int t = dir ? (63 - step) : step;
  const float* gi = (dir ? gi_b : gi_f) + b * 1536;
  const float* gh = (dir ? gh_b : gh_f) + b * 1536;
  float* h = (dir ? h_b : h_f) + b * 512;
  float r = sigm(gi[j] + gh[j]);
  float z = sigm(gi[j + 512] + gh[j + 512]);
  float nn = tanhf(gi[j + 1024] + r * gh[j + 1024]);
  float hprev = h[j];
  float ht = (1.f - z) * nn + z * hprev;
  float m = mask[t * 64 + b];
  float hnew = ht * m + hprev * (1.f - m);
  h[j] = hnew;
  utterin[b * 1024 + dir * 512 + j] = (bf16)hnew;
  enc_out[(size_t)(t * 64 + b) * 1024 + dir * 512 + j] = ht * m;
}

// Utterance GRU with h0 = 0 (so gh = u_bhh): uh = (1-z)*n
__global__ void utter_gate(const float* gi_u, const float* u_bhh,
                           float* h_dec, bf16* hdec_bf, bf16* xcat,
                           float* utter_out) {
  int i = blockIdx.x * blockDim.x + threadIdx.x;   // 64*1024
  if (i >= 64 * 1024) return;
  int j = i & 1023, b = i >> 10;
  const float* gi = gi_u + b * 3072;
  float r = sigm(gi[j] + u_bhh[j]);
  float z = sigm(gi[j + 1024] + u_bhh[j + 1024]);
  float nn = tanhf(gi[j + 2048] + r * u_bhh[j + 2048]);
  float h = (1.f - z) * nn;
  h_dec[i] = h;
  hdec_bf[i] = (bf16)h;
  xcat[b * 2048 + 1024 + j] = (bf16)h;
  utter_out[i] = h;
}

__global__ void dec_gate(const float* gi, const float* gh,
                         float* h_dec, bf16* hdec_bf, bf16* xcat) {
  int i = blockIdx.x * blockDim.x + threadIdx.x;   // 64*1024
  if (i >= 64 * 1024) return;
  int j = i & 1023, b = i >> 10;
  const float* gip = gi + b * 3072;
  const float* ghp = gh + b * 3072;
  float r = sigm(gip[j] + ghp[j]);
  float z = sigm(gip[j + 1024] + ghp[j + 1024]);
  float nn = tanhf(gip[j + 2048] + r * ghp[j + 2048]);
  float hp = h_dec[i];
  float hnew = (1.f - z) * nn + z * hp;
  h_dec[i] = hnew;
  hdec_bf[i] = (bf16)hnew;
  xcat[b * 2048 + 1024 + j] = (bf16)hnew;
}

// Fused attention per batch row b: scores -> softmax -> weighted context.
__global__ void attn_kernel(const float* target, const float* enc_out,
                            bf16* xcat) {
  const int S = 64, B = 64, H = 1024;
  int b = blockIdx.x;
  __shared__ float sc[64];
  int tid = threadIdx.x;
  int lane = tid & 31, wid = tid >> 5;             // 8 waves
  const float* tg = target + (size_t)b * H;
  for (int s = wid; s < S; s += 8) {
    const float* e = enc_out + (size_t)(s * B + b) * H;
    float acc = 0.f;
    for (int d = lane; d < H; d += 32) acc += e[d] * tg[d];
    for (int off = 16; off; off >>= 1) acc += __shfl_xor(acc, off, 32);
    if (lane == 0) sc[s] = acc;
  }
  __syncthreads();
  if (tid == 0) {
    float mx = sc[0];
    for (int s = 1; s < S; ++s) mx = fmaxf(mx, sc[s]);
    float sum = 0.f;
    for (int s = 0; s < S; ++s) { float e = expf(sc[s] - mx); sc[s] = e; sum += e; }
    float inv = 1.f / sum;
    for (int s = 0; s < S; ++s) sc[s] *= inv;
  }
  __syncthreads();
  for (int d = tid; d < H; d += blockDim.x) {
    float acc = 0.f;
    for (int s = 0; s < S; ++s) acc += sc[s] * enc_out[(size_t)(s * B + b) * H + d];
    xcat[b * 2048 + d] = (bf16)acc;
  }
}

// ---------------------------------------------------------------------------
// Host orchestration
// ---------------------------------------------------------------------------
extern "C" void kernel_launch(void* const* d_in, const int* in_sizes, int n_in,
                              void* d_out, int out_size, void* d_ws, size_t ws_size,
                              hipStream_t stream) {
  (void)in_sizes; (void)n_in; (void)out_size; (void)ws_size;
  const int S = 64, T = 48, Bb = 64, E = 512, HH = 512, H = 1024;

  const int*   src      = (const int*)  d_in[0];
  const int*   trg      = (const int*)  d_in[1];
  const float* emb      = (const float*)d_in[2];
  const float* e1_wih   = (const float*)d_in[3];
  const float* e1_whh   = (const float*)d_in[4];
  const float* e1_bih   = (const float*)d_in[5];
  const float* e1_bhh   = (const float*)d_in[6];
  const float* e2_wih   = (const float*)d_in[7];
  const float* e2_whh   = (const float*)d_in[8];
  const float* e2_bih   = (const float*)d_in[9];
  const float* e2_bhh   = (const float*)d_in[10];
  const float* u_wih    = (const float*)d_in[11];
  const float* u_bih    = (const float*)d_in[13];
  const float* u_bhh    = (const float*)d_in[14];
  const float* d_wih    = (const float*)d_in[15];
  const float* d_whh    = (const float*)d_in[16];
  const float* d_bih    = (const float*)d_in[17];
  const float* d_bhh    = (const float*)d_in[18];
  const float* ain_w    = (const float*)d_in[19];
  const float* aout_w   = (const float*)d_in[20];
  float* out = (float*)d_out;

  char* wsb = (char*)d_ws;
  size_t off = 0;
  auto alloc = [&](size_t bytes) -> void* {
    void* p = wsb + off;
    off = (off + bytes + 255) & ~(size_t)255;
    return p;
  };

  // bf16 weight copies
  bf16* e1_wih_bf = (bf16*)alloc((size_t)3 * HH * E * 2);
  bf16* e1_whh_bf = (bf16*)alloc((size_t)3 * HH * HH * 2);
  bf16* e2_wih_bf = (bf16*)alloc((size_t)3 * HH * E * 2);
  bf16* e2_whh_bf = (bf16*)alloc((size_t)3 * HH * HH * 2);
  bf16* u_wih_bf  = (bf16*)alloc((size_t)3 * H * H * 2);
  bf16* d_wih_bf  = (bf16*)alloc((size_t)3 * H * (E + H) * 2);
  bf16* d_whh_bf  = (bf16*)alloc((size_t)3 * H * H * 2);
  bf16* ain_bf    = (bf16*)alloc((size_t)H * H * 2);
  bf16* aout_bf   = (bf16*)alloc((size_t)H * 2 * H * 2);
  // activations
  bf16*  src_emb_bf = (bf16*)alloc((size_t)S * Bb * E * 2);
  bf16*  trg_emb_bf = (bf16*)alloc((size_t)T * Bb * E * 2);
  float* mask       = (float*)alloc((size_t)S * Bb * 4);
  float* h_f        = (float*)alloc((size_t)Bb * HH * 4);
  float* h_b        = (float*)alloc((size_t)Bb * HH * 4);
  bf16*  utterin_bf = (bf16*)alloc((size_t)Bb * H * 2);      // [hf | hb] bf16
  float* g_if = (float*)alloc((size_t)Bb * 3 * HH * 4);
  float* g_hf = (float*)alloc((size_t)Bb * 3 * HH * 4);
  float* g_ib = (float*)alloc((size_t)Bb * 3 * HH * 4);
  float* g_hb = (float*)alloc((size_t)Bb * 3 * HH * 4);
  float* enc_out = (float*)alloc((size_t)S * Bb * H * 4);    // [S,B,2*HH]
  float* gi_u    = (float*)alloc((size_t)Bb * 3 * H * 4);
  float* h_dec   = (float*)alloc((size_t)Bb * H * 4);
  bf16*  hdec_bf = (bf16*)alloc((size_t)Bb * H * 2);
  bf16*  out_bf  = (bf16*)alloc((size_t)Bb * H * 2);
  float* gh_d    = (float*)alloc((size_t)Bb * 3 * H * 4);
  float* gi_full = (float*)alloc((size_t)Bb * 3 * H * 4);
  float* target  = (float*)alloc((size_t)Bb * H * 4);
  bf16*  xcat_bf = (bf16*)alloc((size_t)Bb * 2 * H * 2);

  auto cvt = [&](const float* s, bf16* d, int n) {
    cvt_f32_bf16<<<(n + 255) / 256, 256, 0, stream>>>(s, d, n);
  };
  auto gemm = [&](const bf16* A0, int lda0, const bf16* B0, int ldb0, int K0,
                  const bf16* A1, int lda1, const bf16* B1, int ldb1, int K1,
                  const float* bias, const float* addC, float* C, bf16* Cbf,
                  int M, int N, int act) {
    int waves = (M >> 4) * (N >> 5);
    int blocks = (waves * 32 + 255) / 256;
    gemm_bf16_kernel<<<blocks, 256, 0, stream>>>(A0, lda0, B0, ldb0, K0,
                                                 A1, lda1, B1, ldb1, K1,
                                                 bias, addC, C, Cbf, M, N, act);
  };

  // --- Phase A: weight conversion + embedding gather ---
  cvt(e1_wih, e1_wih_bf, 3 * HH * E);
  cvt(e1_whh, e1_whh_bf, 3 * HH * HH);
  cvt(e2_wih, e2_wih_bf, 3 * HH * E);
  cvt(e2_whh, e2_whh_bf, 3 * HH * HH);
  cvt(u_wih,  u_wih_bf,  3 * H * H);
  cvt(d_wih,  d_wih_bf,  3 * H * (E + H));
  cvt(d_whh,  d_whh_bf,  3 * H * H);
  cvt(ain_w,  ain_bf,    H * H);
  cvt(aout_w, aout_bf,   H * 2 * H);
  embed_gather<<<S * Bb, 128, 0, stream>>>(src, emb, src_emb_bf, mask, E);
  embed_gather<<<T * Bb, 128, 0, stream>>>(trg, emb, trg_emb_bf, nullptr, E);

  // zero initial states
  hipMemsetAsync(h_f, 0, (size_t)Bb * HH * 4, stream);
  hipMemsetAsync(h_b, 0, (size_t)Bb * HH * 4, stream);
  hipMemsetAsync(utterin_bf, 0, (size_t)Bb * H * 2, stream);
  hipMemsetAsync(out_bf, 0, (size_t)Bb * H * 2, stream);

  // --- Phase B: bidirectional encoder, 64 sequential steps ---
  for (int step = 0; step < S; ++step) {
    int tf = step, tb = S - 1 - step;
    Enc4 p;
    p.A[0] = src_emb_bf + (size_t)tf * Bb * E; p.lda[0] = E;
    p.A[1] = utterin_bf;                       p.lda[1] = H;      // hf at cols 0..511
    p.A[2] = src_emb_bf + (size_t)tb * Bb * E; p.lda[2] = E;
    p.A[3] = utterin_bf + HH;                  p.lda[3] = H;      // hb at cols 512..1023
    p.B[0] = e1_wih_bf; p.B[1] = e1_whh_bf; p.B[2] = e2_wih_bf; p.B[3] = e2_whh_bf;
    p.bias[0] = e1_bih; p.bias[1] = e1_bhh; p.bias[2] = e2_bih; p.bias[3] = e2_bhh;
    p.C[0] = g_if; p.C[1] = g_hf; p.C[2] = g_ib; p.C[3] = g_hb;
    enc_step_gemm<<<96, 256, 0, stream>>>(p);                    // 4*192 waves
    enc_gate<<<(2 * 64 * 512) / 256, 256, 0, stream>>>(g_if, g_hf, g_ib, g_hb,
                                                       h_f, h_b, utterin_bf,
                                                       enc_out, mask, step);
  }

  // --- Phase C: utterance GRU cell (h0 = 0) ---
  gemm(utterin_bf, H, u_wih_bf, H, H, nullptr, 0, nullptr, 0, 0,
       u_bih, nullptr, gi_u, nullptr, Bb, 3 * H, 0);
  utter_gate<<<(64 * 1024) / 256, 256, 0, stream>>>(gi_u, u_bhh, h_dec, hdec_bf,
                                                    xcat_bf, out + (size_t)T * Bb * H);

  // --- Phase D: decoder with attention, 48 sequential steps ---
  for (int t = 0; t < T; ++t) {
    // gh = h @ d_whh^T + d_bhh
    gemm(hdec_bf, H, d_whh_bf, H, H, nullptr, 0, nullptr, 0, 0,
         d_bhh, nullptr, gh_d, nullptr, Bb, 3 * H, 0);
    // gi = [emb_t | out] @ d_wih^T + d_bih   (dual-K fused concat)
    gemm(trg_emb_bf + (size_t)t * Bb * E, E, d_wih_bf, E + H, E,
         out_bf, H, d_wih_bf + E, E + H, H,
         d_bih, nullptr, gi_full, nullptr, Bb, 3 * H, 0);
    dec_gate<<<(64 * 1024) / 256, 256, 0, stream>>>(gi_full, gh_d, h_dec, hdec_bf, xcat_bf);
    // target = h @ attn_in_w^T
    gemm(hdec_bf, H, ain_bf, H, H, nullptr, 0, nullptr, 0, 0,
         nullptr, nullptr, target, nullptr, Bb, H, 0);
    attn_kernel<<<64, 256, 0, stream>>>(target, enc_out, xcat_bf);
    // out = tanh([wctx | h] @ attn_out_w^T) -> d_out[t] (f32) + out_bf (bf16 state)
    gemm(xcat_bf, 2 * H, aout_bf, 2 * H, 2 * H, nullptr, 0, nullptr, 0, 0,
         nullptr, nullptr, out + (size_t)t * Bb * H, out_bf, Bb, H, 1);
  }
}
